// GNOME_42588895707869
// MI455X (gfx1250) — compile-verified, matrix operand order checked by reading, not run on hardware
//
#include <hip/hip_runtime.h>
#include <math.h>

// ---------------------------------------------------------------------------
// Types for CDNA5 WMMA (gfx1250, wave32)
// ---------------------------------------------------------------------------
typedef __attribute__((ext_vector_type(16))) __bf16 v16bf;
typedef __attribute__((ext_vector_type(4)))  __bf16 v4bf;
typedef __attribute__((ext_vector_type(4)))  float  v4f;
typedef __attribute__((ext_vector_type(8)))  float  v8f;

__device__ inline v8f wmma_bf16(v16bf a, v16bf b, v8f c) {
    return __builtin_amdgcn_wmma_f32_16x16x32_bf16(false, a, false, b,
                                                   (short)0, c, false, false);
}

// Fragment element map (ISA 7.12.2, 16-bit A 16x32 / B 32x16):
//   lane half h = lane>>4;  element e in 0..15  <->  k = (e<8) ? h*8+e : 16+h*8+(e-8)
//   inverse: half = (k>>3)&1 ; e = (k&7) + 8*(k>>4)

// ---------------------------------------------------------------------------
// Pack fp32 B (KxN row-major, or NxK row-major if transB) into bf16
// fragment-blocked layout: [nblk][kblk][lane][16], one 16x16(x32) fragment
// slice per (nblk,kblk,lane). GEMM reads B fragments as contiguous 32B/lane.
// ---------------------------------------------------------------------------
__global__ void pack_b_bf16(const float* __restrict__ B, int ldb, int transB,
                            int K, int N, __bf16* __restrict__ out)
{
    int slot = blockIdx.x * blockDim.x + threadIdx.x;  // (nblk,kblk,lane)
    int kblocks = K >> 5;
    int total = (N >> 4) * kblocks * 32;
    if (slot >= total) return;
    int lane = slot & 31;
    int kblk = (slot >> 5) % kblocks;
    int nblk = (slot >> 5) / kblocks;
    int n = (nblk << 4) + (lane & 15);
    int half = lane >> 4;
    v16bf v;
#pragma unroll
    for (int e = 0; e < 16; ++e) {
        int kl = (e < 8) ? (half * 8 + e) : (8 + half * 8 + e); // = 16+h*8+(e-8)
        int k = (kblk << 5) + kl;
        float f = transB ? B[(size_t)n * ldb + k] : B[(size_t)k * ldb + n];
        v[e] = (__bf16)f;
    }
    *(v16bf*)(out + (size_t)slot * 16) = v;
}

// ---------------------------------------------------------------------------
// bf16-WMMA GEMM: C[M,N] = act( A[M,K] * Bpacked + bias + res ), fp32 A/C.
//   - A staged to LDS in fragment layout (v4f load -> cvt -> ds_store_b64).
//   - B read directly from packed global (2x global_load_b128 per fragment).
//   - feat_out: optional pre-activation writeback (GINE residual 'feat').
//   - na/nb non-null => cdist epilogue sqrt(max(na[i]+nb[j]-2*acc, 1e-12)).
// Block: 256 threads / 8 waves; block tile 64(M) x 128(N); wave tile 32x32;
// K step 32. Requires M%64==0, N%128==0, K%32==0 (true for all shapes here).
// ---------------------------------------------------------------------------
__global__ __launch_bounds__(256) void gemm_bf16_wmma(
    const float* __restrict__ A, int lda,
    const __bf16* __restrict__ Bp,
    const float* __restrict__ bias,
    const float* __restrict__ res, int ldres,
    float* __restrict__ feat_out, int ldfeat,
    float* __restrict__ C, int ldc,
    int M, int N, int K, int act,
    const float* __restrict__ na, const float* __restrict__ nb)
{
    __shared__ __bf16 As[4 * 32 * 16];   // 4 M-strips x 32 lane-slots x 16 elems

    const int kblocks = K >> 5;
    const int bm = blockIdx.y * 64;
    const int bn = blockIdx.x * 128;
    const int tid  = threadIdx.x;
    const int lane = tid & 31;
    const int wave = tid >> 5;
    const int wm = (wave >> 2) << 5;     // 0 or 32
    const int wn = (wave & 3) << 5;      // 0,32,64,96

    // A-staging addressing (constant per thread): 512 vec4 slots, 2/thread.
    // slot v: row r = v>>3, k-offset c4 = (v&7)*4; fragment position:
    //   strip=r>>4, lane-slot = 16*((c4>>3)&1) + (r&15), e0 = (c4&7)+8*(c4>>4)
    int v0 = tid, v1 = tid + 256;
    int r0 = v0 >> 3, c40 = (v0 & 7) << 2;
    int r1 = v1 >> 3, c41 = (v1 & 7) << 2;
    int dst0 = (r0 >> 4) * 512 + ((((c40 >> 3) & 1) << 4) + (r0 & 15)) * 16
             + ((c40 & 7) + ((c40 >> 4) << 3));
    int dst1 = (r1 >> 4) * 512 + ((((c41 >> 3) & 1) << 4) + (r1 & 15)) * 16
             + ((c41 & 7) + ((c41 >> 4) << 3));

    v8f acc[2][2] = {};

    const __bf16* bbase = Bp + ((size_t)((bn + wn) >> 4) * kblocks) * 512
                             + (size_t)lane * 16;
    const size_t bstride_n = (size_t)kblocks * 512;   // next nblk

    for (int kb = 0; kb < kblocks; ++kb) {
        // ---- stage A tile (64x32) into fragment-layout LDS ----
        {
            v4f a4 = *(const v4f*)(A + (size_t)(bm + r0) * lda + (kb << 5) + c40);
            v4bf h4; h4[0] = (__bf16)a4[0]; h4[1] = (__bf16)a4[1];
                     h4[2] = (__bf16)a4[2]; h4[3] = (__bf16)a4[3];
            *(v4bf*)(As + dst0) = h4;
        }
        {
            v4f a4 = *(const v4f*)(A + (size_t)(bm + r1) * lda + (kb << 5) + c41);
            v4bf h4; h4[0] = (__bf16)a4[0]; h4[1] = (__bf16)a4[1];
                     h4[2] = (__bf16)a4[2]; h4[3] = (__bf16)a4[3];
            *(v4bf*)(As + dst1) = h4;
        }
        __syncthreads();

        // ---- fragments: A from LDS (2x ds_load_b128 each), B from global ----
        v16bf af0 = *(const v16bf*)(As + ((wm >> 4) + 0) * 512 + lane * 16);
        v16bf af1 = *(const v16bf*)(As + ((wm >> 4) + 1) * 512 + lane * 16);
        const __bf16* bb = bbase + (size_t)kb * 512;
        v16bf bf0 = *(const v16bf*)(bb);
        v16bf bf1 = *(const v16bf*)(bb + bstride_n);

        acc[0][0] = wmma_bf16(af0, bf0, acc[0][0]);
        acc[0][1] = wmma_bf16(af0, bf1, acc[0][1]);
        acc[1][0] = wmma_bf16(af1, bf0, acc[1][0]);
        acc[1][1] = wmma_bf16(af1, bf1, acc[1][1]);
        __syncthreads();
    }

    // ---- epilogue. C/D layout: VGPR r -> M = r + 8*(lane>=16); N = lane&15.
    const int nloc = lane & 15;
    const int mb   = (lane >> 4) << 3;
#pragma unroll
    for (int s = 0; s < 2; ++s) {
#pragma unroll
        for (int j = 0; j < 2; ++j) {
            int col = bn + wn + (j << 4) + nloc;
#pragma unroll
            for (int r = 0; r < 8; ++r) {
                int row = bm + wm + (s << 4) + mb + r;
                float vv = acc[s][j][r];
                if (na) {
                    float d2 = na[row] + nb[col] - 2.0f * vv;
                    C[(size_t)row * ldc + col] = sqrtf(fmaxf(d2, 1e-12f));
                } else {
                    if (bias) vv += bias[col];
                    if (res)  vv += res[(size_t)row * ldres + col];
                    if (feat_out) feat_out[(size_t)row * ldfeat + col] = vv;
                    if (act) vv = fmaxf(vv, 0.0f);
                    C[(size_t)row * ldc + col] = vv;
                }
            }
        }
    }
}

// ---------------------------------------------------------------------------
// e = e_features @ W_edge + b_edge          (E x 8) x (8 x 128)
// ---------------------------------------------------------------------------
__global__ __launch_bounds__(128) void edge_linear(
    const float* __restrict__ ef, const float* __restrict__ W,
    const float* __restrict__ b, float* __restrict__ out)
{
    __shared__ float s[8];
    int e = blockIdx.x, h = threadIdx.x;
    if (h < 8) s[h] = ef[(size_t)e * 8 + h];
    __syncthreads();
    float acc = b[h];
#pragma unroll
    for (int k = 0; k < 8; ++k) acc += s[k] * W[k * 128 + h];
    out[(size_t)e * 128 + h] = acc;
}

// ---------------------------------------------------------------------------
// x = concat(features[48], RW[16]) @ W_pre + b_pre     (N x 64) x (64 x 128)
// ---------------------------------------------------------------------------
__global__ __launch_bounds__(128) void pre_linear(
    const float* __restrict__ feat, const float* __restrict__ rw,
    const float* __restrict__ W, const float* __restrict__ b,
    float* __restrict__ out)
{
    __shared__ float s[64];
    int n = blockIdx.x, h = threadIdx.x;
    if (h < 48)      s[h] = feat[(size_t)n * 48 + h];
    else if (h < 64) s[h] = rw[(size_t)n * 16 + (h - 48)];
    __syncthreads();
    float acc = b[h];
#pragma unroll
    for (int k = 0; k < 64; ++k) acc += s[k] * W[k * 128 + h];
    out[(size_t)n * 128 + h] = acc;
}

// agg[n, 0:128] = in[n*ldin + 0:128]   (init agg = x, folds (1+eps)*x_i term)
__global__ void copy_slice(const float* __restrict__ in, int ldin,
                           float* __restrict__ out, int rows)
{
    size_t idx = (size_t)blockIdx.x * blockDim.x + threadIdx.x;
    if (idx >= (size_t)rows * 128) return;
    int r = (int)(idx >> 7), h = (int)(idx & 127);
    out[idx] = in[(size_t)r * ldin + h];
}

// agg[dst] += relu(x[src] + e)   (segment_sum via f32 atomics to L2)
__global__ void scatter_msg(const float* __restrict__ x, int ldx,
                            const float* __restrict__ ebuf,
                            const int* __restrict__ src,
                            const int* __restrict__ dst,
                            float* __restrict__ agg, int E)
{
    size_t idx = (size_t)blockIdx.x * blockDim.x + threadIdx.x;
    if (idx >= (size_t)E * 128) return;
    int e = (int)(idx >> 7), h = (int)(idx & 127);
    int s = src[e], d = dst[e];
    float m = x[(size_t)s * ldx + h] + ebuf[idx];
    m = fmaxf(m, 0.0f);
    atomicAdd(&agg[(size_t)d * 128 + h], m);
}

// out[row] = sum_c m[row,c]^2
__global__ __launch_bounds__(256) void row_norm2(
    const float* __restrict__ m, int ld, int cols, float* __restrict__ out)
{
    __shared__ float red[256];
    int row = blockIdx.x;
    float s = 0.0f;
    for (int c = threadIdx.x; c < cols; c += 256) {
        float v = m[(size_t)row * ld + c];
        s += v * v;
    }
    red[threadIdx.x] = s;
    __syncthreads();
    for (int o = 128; o > 0; o >>= 1) {
        if ((int)threadIdx.x < o) red[threadIdx.x] += red[threadIdx.x + o];
        __syncthreads();
    }
    if (threadIdx.x == 0) out[row] = red[0];
}

// ---------------------------------------------------------------------------
extern "C" void kernel_launch(void* const* d_in, const int* in_sizes, int n_in,
                              void* d_out, int out_size, void* d_ws, size_t ws_size,
                              hipStream_t stream)
{
    (void)in_sizes; (void)n_in; (void)out_size; (void)ws_size;
    const int N = 8192, E = 131072, H = 128, L = 6, CAT = 768, HID = 1536;

    const int*   ei[2]  = { (const int*)d_in[0],   (const int*)d_in[1]   };
    const float* fea[2] = { (const float*)d_in[2], (const float*)d_in[3] };
    const float* ef[2]  = { (const float*)d_in[4], (const float*)d_in[5] };
    const float* rw[2]  = { (const float*)d_in[6], (const float*)d_in[7] };
    const float* W_edge = (const float*)d_in[8];
    const float* b_edge = (const float*)d_in[9];
    const float* W_pre  = (const float*)d_in[10];
    const float* b_pre  = (const float*)d_in[11];
    const float* gw1    = (const float*)d_in[12];
    const float* gb1    = (const float*)d_in[13];
    const float* gw2    = (const float*)d_in[14];
    const float* gb2    = (const float*)d_in[15];
    const float* W_out1 = (const float*)d_in[16];
    const float* b_out1 = (const float*)d_in[17];
    const float* W_out2 = (const float*)d_in[18];
    const float* b_out2 = (const float*)d_in[19];

    // workspace carve-out
    char* cur = (char*)d_ws;
    auto alloc = [&](size_t bytes) -> char* {
        char* r = cur;
        cur += (bytes + 255) & ~(size_t)255;
        return r;
    };
    float* ebuf[2] = { (float*)alloc((size_t)E * H * 4), (float*)alloc((size_t)E * H * 4) };
    float* xg[2]   = { (float*)alloc((size_t)N * H * 4), (float*)alloc((size_t)N * H * 4) };
    float* agg     = (float*)alloc((size_t)N * H * 4);
    float* tbuf    = (float*)alloc((size_t)N * H * 4);
    float* featbuf = (float*)alloc((size_t)N * H * 4);
    float* ne[2]   = { (float*)alloc((size_t)N * CAT * 4), (float*)alloc((size_t)N * CAT * 4) };
    float* t1      = (float*)alloc((size_t)N * HID * 4);
    float* mm[2]   = { (float*)alloc((size_t)N * CAT * 4), (float*)alloc((size_t)N * CAT * 4) };
    float* nrm[2]  = { (float*)alloc((size_t)N * 4), (float*)alloc((size_t)N * 4) };
    // packed bf16 B operands
    const size_t pw_sz = (size_t)H * H;                // per-layer 128x128
    __bf16* pw1  = (__bf16*)alloc(6 * pw_sz * 2);
    __bf16* pw2  = (__bf16*)alloc(6 * pw_sz * 2);
    __bf16* pWo1 = (__bf16*)alloc((size_t)CAT * HID * 2);
    __bf16* pWo2 = (__bf16*)alloc((size_t)HID * CAT * 2);
    __bf16* pm2  = (__bf16*)alloc((size_t)CAT * N * 2);

    auto pack = [&](const float* B, int ldb, int trB, int K, int Ncols, __bf16* out) {
        int slots = (Ncols >> 4) * (K >> 5) * 32;
        pack_b_bf16<<<(slots + 255) / 256, 256, 0, stream>>>(B, ldb, trB, K, Ncols, out);
    };
    auto gemm = [&](const float* A, int lda, const __bf16* Bp,
                    const float* bias, const float* res, int ldres,
                    float* feat, int ldfeat, float* C, int ldc,
                    int M, int Ncols, int K, int act,
                    const float* pna, const float* pnb) {
        dim3 grid(Ncols / 128, M / 64);
        gemm_bf16_wmma<<<grid, 256, 0, stream>>>(A, lda, Bp, bias, res, ldres,
                                                 feat, ldfeat, C, ldc,
                                                 M, Ncols, K, act, pna, pnb);
    };

    // pack all weight matrices (once per launch; deterministic)
    for (int i = 0; i < L; ++i) {
        pack(gw1 + (size_t)i * H * H, H, 0, H, H, pw1 + (size_t)i * pw_sz);
        pack(gw2 + (size_t)i * H * H, H, 0, H, H, pw2 + (size_t)i * pw_sz);
    }
    pack(W_out1, HID, 0, CAT, HID, pWo1);
    pack(W_out2, CAT, 0, HID, CAT, pWo2);

    const int scat_blocks = (int)(((size_t)E * 128 + 255) / 256);
    const int copy_blocks = (int)(((size_t)N * 128 + 255) / 256);

    for (int g = 0; g < 2; ++g) {
        edge_linear<<<E, 128, 0, stream>>>(ef[g], W_edge, b_edge, ebuf[g]);
        pre_linear<<<N, 128, 0, stream>>>(fea[g], rw[g], W_pre, b_pre, xg[g]);

        const int* src = ei[g];          // edge_index[0]
        const int* dst = ei[g] + E;      // edge_index[1]

        for (int i = 0; i < L; ++i) {
            const float* xcur = (i == 0) ? xg[g] : (ne[g] + (size_t)(i - 1) * H);
            int ldx = (i == 0) ? H : CAT;

            copy_slice<<<copy_blocks, 256, 0, stream>>>(xcur, ldx, agg, N);
            scatter_msg<<<scat_blocks, 256, 0, stream>>>(xcur, ldx, ebuf[g],
                                                         src, dst, agg, E);

            // t = relu(agg @ W1[i] + b1[i])
            gemm(agg, H, pw1 + (size_t)i * pw_sz, gb1 + (size_t)i * H,
                 nullptr, 0, nullptr, 0, tbuf, H, N, H, H, 1, nullptr, nullptr);

            // h = t @ W2[i] + b2[i]  (+ 'feat' residual on odd non-last layers)
            const float* res = nullptr;
            float* fout = nullptr;
            if (i == 1) { res = xg[g];   fout = featbuf; }
            if (i == 3) { res = featbuf; fout = featbuf; }
            gemm(tbuf, H, pw2 + (size_t)i * pw_sz, gb2 + (size_t)i * H,
                 res, H, fout, H, ne[g] + (size_t)i * H, CAT,
                 N, H, H, /*relu*/1, nullptr, nullptr);
        }

        // output MLP
        gemm(ne[g], CAT, pWo1, b_out1, nullptr, 0, nullptr, 0,
             t1, HID, N, HID, CAT, /*relu*/1, nullptr, nullptr);
        gemm(t1, HID, pWo2, b_out2, nullptr, 0, nullptr, 0,
             mm[g], CAT, N, CAT, HID, /*none*/0, nullptr, nullptr);

        row_norm2<<<N, 256, 0, stream>>>(mm[g], CAT, CAT, nrm[g]);
    }

    // pack m2^T, then out[i,j] = sqrt(max(|m1_i|^2+|m2_j|^2-2 m1_i.m2_j, 1e-12))
    pack(mm[1], CAT, /*trans*/1, CAT, N, pm2);
    gemm(mm[0], CAT, pm2, nullptr, nullptr, 0, nullptr, 0,
         (float*)d_out, N, N, N, CAT, 0, nrm[0], nrm[1]);
}
